// DSALite_84559316123679
// MI455X (gfx1250) — matrix operand order repaired and checked by reading, not compiled, not run on hardware
//
#include <hip/hip_runtime.h>

// ---------------------------------------------------------------------------
// DSA-lite attention for MI455X (gfx1250), wave32, WMMA f16 path.
//
// Phase 1: per (head, ds-row): fp32 ds scores (512 cols), exact 128th-largest
//          threshold via bitonic sort in LDS, writes D = s_ds - thr to d_ws.
// Phase 2: flash attention over full rows with bias
//          (1 - causal*sigmoid(10*u_i*D)) * NEG_INF, f16 WMMA for QK^T and
//          P*V with fp32 accumulation. Causal blocks first, then a
//          workgroup-uniform vote decides whether the (numerically tied)
//          non-causal tail must be processed to match reference semantics.
//
// Round-4: revert to the single-buffered, unconditionally-rescaled round-2
// structure (double-buffering was defeated by in-order DScnt and the
// conditional rescale forced accumulator copies + WMMA->VALU hazard NOPs);
// keep SCALE folded into the f16 Q fragments and the fused K/V staging loop.
// ---------------------------------------------------------------------------

typedef __attribute__((ext_vector_type(16))) _Float16 v16h;
typedef __attribute__((ext_vector_type(8)))  float    v8f;

#define TT     4096
#define HD     128
#define NH     4
#define TDS    512            // TT / 8 (STRIDE)
#define KSEL   128            // 128th largest of 512 ds scores
#define NEGINF (-1000000000.0f)
#define SCALE  0.08838834764831845f   // 128^-0.5

// --- 16-lane-row butterfly reductions, result replicated in all 16 lanes.
__device__ __forceinline__ float dpp_rowmax16(float x) {
  float y;
  y = __int_as_float(__builtin_amdgcn_update_dpp(0, __float_as_int(x), 0xB1, 0xf, 0xf, true));
  x = fmaxf(x, y);
  y = __int_as_float(__builtin_amdgcn_update_dpp(0, __float_as_int(x), 0x4E, 0xf, 0xf, true));
  x = fmaxf(x, y);
  y = __int_as_float(__builtin_amdgcn_update_dpp(0, __float_as_int(x), 0x141, 0xf, 0xf, true));
  x = fmaxf(x, y);
  y = __int_as_float(__builtin_amdgcn_update_dpp(0, __float_as_int(x), 0x140, 0xf, 0xf, true));
  x = fmaxf(x, y);
  return x;
}
__device__ __forceinline__ float dpp_rowsum16(float x) {
  float y;
  y = __int_as_float(__builtin_amdgcn_update_dpp(0, __float_as_int(x), 0xB1, 0xf, 0xf, true));
  x += y;
  y = __int_as_float(__builtin_amdgcn_update_dpp(0, __float_as_int(x), 0x4E, 0xf, 0xf, true));
  x += y;
  y = __int_as_float(__builtin_amdgcn_update_dpp(0, __float_as_int(x), 0x141, 0xf, 0xf, true));
  x += y;
  y = __int_as_float(__builtin_amdgcn_update_dpp(0, __float_as_int(x), 0x140, 0xf, 0xf, true));
  x += y;
  return x;
}

// ---------------------------------------------------------------------------
// Phase 1: D[h][rds][c] = S_ds - thr(rds)
// ---------------------------------------------------------------------------
__global__ __launch_bounds__(128) void dsal_phase1(const float* __restrict__ Q,
                                                   const float* __restrict__ K,
                                                   float* __restrict__ D) {
  __shared__ float qv[HD];
  __shared__ float sc[TDS];
  __shared__ float srt[TDS];
  const int tid = threadIdx.x;
  const int rds = blockIdx.x;      // 0..511
  const int h   = blockIdx.y;      // 0..3

  const float* qrow = Q + ((size_t)(h * TT + rds * 8)) * HD;
  qv[tid] = qrow[tid];             // blockDim == HD == 128
  __syncthreads();

  for (int c = tid; c < TDS; c += 128) {
    const float* krow = K + ((size_t)(h * TT + c * 8)) * HD;
    float acc = 0.f;
    for (int d = 0; d < HD; ++d) acc += qv[d] * krow[d];
    float s = acc * SCALE;
    sc[c] = s;
    srt[c] = s;
  }
  __syncthreads();

  // bitonic ascending sort of srt[0..511]
  for (int k = 2; k <= TDS; k <<= 1) {
    for (int j = k >> 1; j > 0; j >>= 1) {
      for (int i = tid; i < TDS; i += 128) {
        int ixj = i ^ j;
        if (ixj > i) {
          bool up = ((i & k) == 0);
          float a = srt[i], b = srt[ixj];
          if ((a > b) == up) { srt[i] = b; srt[ixj] = a; }
        }
      }
      __syncthreads();
    }
  }
  float thr = srt[TDS - KSEL];     // 128th largest
  for (int c = tid; c < TDS; c += 128)
    D[(((size_t)h * TDS) + rds) * TDS + c] = sc[c] - thr;
}

// ---------------------------------------------------------------------------
// Phase 2: flash attention. 128 threads = 4 waves; 16 query rows per wave.
// ---------------------------------------------------------------------------
#define KST 144   // halves per K-tile row   (288B, 32B aligned)
#define VST 80    // halves per Vt row       (160B, 32B aligned)
#define PST 72    // halves per P row        (144B, 16B aligned)

__global__ __launch_bounds__(128) void dsal_phase2(
    const float* __restrict__ Q, const float* __restrict__ K,
    const float* __restrict__ V, const float* __restrict__ U,
    const float* __restrict__ D, float* __restrict__ Out) {
  __shared__ _Float16 Ksh[64 * KST] __attribute__((aligned(32)));
  __shared__ _Float16 Vt[HD * VST]  __attribute__((aligned(32)));
  __shared__ _Float16 Psh[4][16 * PST] __attribute__((aligned(32)));
  __shared__ int sflag;

  const int tid  = threadIdx.x;
  const int lane = tid & 31;
  const int wave = tid >> 5;         // 0..3
  const int half = lane >> 4;        // 0/1 (upper lane group)
  const int l16  = lane & 15;
  const int h    = blockIdx.y;
  const int rb   = blockIdx.x;       // 64-row block index
  const int wRow = rb * 64 + wave * 16;

  // --- Q A-fragments (16-bit A 16x32, contiguous k = 16*half + s), SCALE pre-folded
  v16h aQ[4];
  {
    const float* qrow = Q + ((size_t)(h * TT + wRow + l16)) * HD;
    #pragma unroll
    for (int t = 0; t < 4; ++t) {
      int base = t * 32 + (half ? 16 : 0);
      v16h a;
      #pragma unroll
      for (int j = 0; j < 8; ++j) a[j] = (_Float16)(qrow[base + j] * SCALE);
      #pragma unroll
      for (int j = 0; j < 8; ++j) a[8 + j] = (_Float16)(qrow[base + 8 + j] * SCALE);
      aQ[t] = a;
    }
  }

  // --- per-row state (row = wRow + g + 8*half; replicated across 16 lanes)
  float m_[8], l_[8], us[8];
  v8f Oc[8];
  #pragma unroll
  for (int g = 0; g < 8; ++g) {
    int row = wRow + g + 8 * half;
    float uu = fminf(fmaxf(U[row], 0.f), 1.f);
    us[g] = 10.f * (1.f + uu);       // sharpness * u_scale, premultiplied
    m_[g] = -3.0e38f;
    l_[g] = 0.f;
    Oc[g] = {};
  }
  const float* Drow = D + (((size_t)h * TDS) + (wRow >> 3) + half) * TDS;

  for (int pass = 0; pass < 2; ++pass) {
    const int jStart = (pass == 0) ? 0 : rb + 1;
    const int jStop  = (pass == 0) ? rb + 1 : TT / 64;

    for (int jB = jStart; jB < jStop; ++jB) {
      const int jBase = jB * 64;
      __syncthreads();
      // fused cooperative stage: K (row-major f16) and V (transposed f16)
      for (int i = tid; i < 64 * 32; i += 128) {
        int r = i >> 5, c4 = i & 31;
        size_t rowOff = ((size_t)(h * TT + jBase + r)) * HD;
        float4 kv = ((const float4*)(K + rowOff))[c4];
        float4 vv = ((const float4*)(V + rowOff))[c4];
        _Float16* d = &Ksh[r * KST + c4 * 4];
        d[0] = (_Float16)kv.x; d[1] = (_Float16)kv.y;
        d[2] = (_Float16)kv.z; d[3] = (_Float16)kv.w;
        _Float16* vt = &Vt[(c4 * 4) * VST + r];
        vt[0 * VST] = (_Float16)vv.x;
        vt[1 * VST] = (_Float16)vv.y;
        vt[2 * VST] = (_Float16)vv.z;
        vt[3 * VST] = (_Float16)vv.w;
      }
      __syncthreads();

      // --- S = Q K^T : 4 column tiles x 4 K-steps of v_wmma_f32_16x16x32_f16
      v8f S[4];
      #pragma unroll
      for (int nt = 0; nt < 4; ++nt) {
        v8f c = {};
        #pragma unroll
        for (int t = 0; t < 4; ++t) {
          const _Float16* bp = &Ksh[(nt * 16 + l16) * KST + t * 32 + (half ? 16 : 0)];
          v16h b = *(const v16h*)bp;
          c = __builtin_amdgcn_wmma_f32_16x16x32_f16(false, aQ[t], false, b,
                                                     (short)0, c, false, false);
        }
        S[nt] = c;
      }

      // --- bias: (1 - causal*sigmoid(us*D)) * NEG_INF, fp32 add (ref rounding)
      #pragma unroll
      for (int nt = 0; nt < 4; ++nt) {
        int col = jBase + nt * 16 + l16;
        float Dv = Drow[col >> 3];
        #pragma unroll
        for (int g = 0; g < 8; ++g) {
          int row = wRow + g + 8 * half;
          float sig = (col <= row)
                        ? __builtin_amdgcn_rcpf(1.f + __expf(-us[g] * Dv))
                        : 0.f;
          S[nt][g] = S[nt][g] + (1.f - sig) * NEGINF;
        }
      }

      // --- online softmax: row max (reduce over nt, then DPP over 16 lanes)
      float al[8];
      #pragma unroll
      for (int g = 0; g < 8; ++g) {
        float r = S[0][g];
        r = fmaxf(r, S[1][g]); r = fmaxf(r, S[2][g]); r = fmaxf(r, S[3][g]);
        r = dpp_rowmax16(r);
        float mn = fmaxf(m_[g], r);
        al[g] = __expf(m_[g] - mn);
        m_[g] = mn;
      }

      // --- P = exp(S - m), row sums, stage P (f16) to wave-private LDS
      float rs[8];
      #pragma unroll
      for (int g = 0; g < 8; ++g) rs[g] = 0.f;
      _Float16* pw = &Psh[wave][0];
      #pragma unroll
      for (int nt = 0; nt < 4; ++nt) {
        #pragma unroll
        for (int g = 0; g < 8; ++g) {
          float p = __expf(S[nt][g] - m_[g]);
          rs[g] += p;
          pw[(g + 8 * half) * PST + nt * 16 + l16] = (_Float16)p;
        }
      }
      #pragma unroll
      for (int g = 0; g < 8; ++g)
        l_[g] = l_[g] * al[g] + dpp_rowsum16(rs[g]);
      #pragma unroll
      for (int dt = 0; dt < 8; ++dt)
        #pragma unroll
        for (int g = 0; g < 8; ++g) Oc[dt][g] *= al[g];

      // --- O += P * V (LDS ops are in-order within a wave; no explicit wait)
      v16h aP[2];
      #pragma unroll
      for (int jt = 0; jt < 2; ++jt) {
        const _Float16* ap = &Psh[wave][l16 * PST + jt * 32 + (half ? 16 : 0)];
        v16h a;
        #pragma unroll
        for (int j = 0; j < 8; ++j) { a[j] = ap[j]; a[8 + j] = ap[8 + j]; }
        aP[jt] = a;
      }
      #pragma unroll
      for (int dt = 0; dt < 8; ++dt) {
        v8f c = Oc[dt];
        #pragma unroll
        for (int jt = 0; jt < 2; ++jt) {
          const _Float16* bp = &Vt[(dt * 16 + l16) * VST + jt * 32 + (half ? 16 : 0)];
          v16h b = *(const v16h*)bp;
          c = __builtin_amdgcn_wmma_f32_16x16x32_f16(false, aP[jt], false, b,
                                                     (short)0, c, false, false);
        }
        Oc[dt] = c;
      }
    }

    if (pass == 0) {
      // Skip non-causal tail only if every row in the workgroup has a score
      // far enough above NEG_INF that exactly-NEG_INF ties cannot contribute.
      int ok = 1;
      #pragma unroll
      for (int g = 0; g < 8; ++g) ok &= (m_[g] > -9.998e8f) ? 1 : 0;
      ok = __all(ok);
      if (tid == 0) sflag = 1;
      __syncthreads();
      if (lane == 0 && !ok) atomicAnd(&sflag, 0);
      __syncthreads();
      if (sflag) break;
    }
  }

  // --- finalize: O /= l, store fp32
  #pragma unroll
  for (int g = 0; g < 8; ++g) l_[g] = __builtin_amdgcn_rcpf(l_[g]);
  #pragma unroll
  for (int dt = 0; dt < 8; ++dt) {
    #pragma unroll
    for (int g = 0; g < 8; ++g) {
      int row = wRow + g + 8 * half;
      Out[((size_t)(h * TT + row)) * HD + dt * 16 + l16] = Oc[dt][g] * l_[g];
    }
  }
}

// ---------------------------------------------------------------------------
extern "C" void kernel_launch(void* const* d_in, const int* in_sizes, int n_in,
                              void* d_out, int out_size, void* d_ws, size_t ws_size,
                              hipStream_t stream) {
  (void)in_sizes; (void)n_in; (void)out_size; (void)ws_size;
  const float* Q = (const float*)d_in[0];
  const float* K = (const float*)d_in[1];
  const float* V = (const float*)d_in[2];
  const float* U = (const float*)d_in[3];
  float* D   = (float*)d_ws;            // NH*512*512 floats = 4 MB scratch
  float* Out = (float*)d_out;

  dim3 g1(TDS, NH);
  dsal_phase1<<<g1, 128, 0, stream>>>(Q, K, D);
  dim3 g2(TT / 64, NH);
  dsal_phase2<<<g2, 128, 0, stream>>>(Q, K, V, U, D, Out);
}
